// GlobalAttention3D_65292092834184
// MI455X (gfx1250) — matrix operand order repaired and checked by reading, not compiled, not run on hardware
//
#include <hip/hip_runtime.h>
#include <math.h>

typedef __bf16 v16bf __attribute__((ext_vector_type(16)));
typedef __bf16 v8bf  __attribute__((ext_vector_type(8)));
typedef float  v8f   __attribute__((ext_vector_type(8)));

#define C_DIM 256
#define HEADS 8
#define HD 32
#define NVOX 4096
#define BATCH 2
#define BN (BATCH * NVOX)   // 8192
#define GROUPS 8

static __device__ __forceinline__ v16bf cat8(v8bf lo, v8bf hi) {
    return __builtin_shufflevector(lo, hi, 0,1,2,3,4,5,6,7,8,9,10,11,12,13,14,15);
}

static __device__ __forceinline__ v8f wmma_bf16(v16bf a, v16bf b, v8f c) {
    return __builtin_amdgcn_wmma_f32_16x16x32_bf16(false, a, false, b, (short)0, c, false, false);
}

// ---------------- GroupNorm stats: one block per (b,g) ----------------
__global__ void gn_stats_kernel(const float* __restrict__ x, float* __restrict__ stats) {
    const int bg = blockIdx.x;                       // b*8+g  (group channels contiguous)
    const float* base = x + (size_t)bg * 32 * NVOX;  // 32 ch * 4096 vox contiguous
    float s = 0.f, ss = 0.f;
    for (int i = threadIdx.x; i < 32 * NVOX; i += blockDim.x) {
        float v = base[i]; s += v; ss += v * v;
    }
    __shared__ float sh[256], sh2[256];
    sh[threadIdx.x] = s; sh2[threadIdx.x] = ss;
    __syncthreads();
    for (int off = 128; off > 0; off >>= 1) {
        if (threadIdx.x < off) { sh[threadIdx.x] += sh[threadIdx.x + off]; sh2[threadIdx.x] += sh2[threadIdx.x + off]; }
        __syncthreads();
    }
    if (threadIdx.x == 0) {
        const float inv_n = 1.0f / (32.0f * NVOX);
        float mu  = sh[0] * inv_n;
        float var = sh2[0] * inv_n - mu * mu;
        stats[bg * 2 + 0] = mu;
        stats[bg * 2 + 1] = rsqrtf(var + 1e-5f);
    }
}

// ---------------- GroupNorm apply + transpose to [b][n][c] bf16 ----------------
__global__ void gn_apply_kernel(const float* __restrict__ x, const float* __restrict__ stats,
                                const float* __restrict__ gw, const float* __restrict__ gb,
                                __bf16* __restrict__ xnT) {
    int idx = blockIdx.x * 256 + threadIdx.x;        // [b][n][c] flat
    int c = idx & 255;
    int n = (idx >> 8) & 4095;
    int b = idx >> 20;
    int bg = b * GROUPS + (c >> 5);
    float mu = stats[bg * 2 + 0], rstd = stats[bg * 2 + 1];
    float v = (x[((size_t)(b * C_DIM + c)) * NVOX + n] - mu) * rstd * gw[c] + gb[c];
    xnT[idx] = (__bf16)v;
}

// ---------------- f32 -> bf16 pack (weights) ----------------
__global__ void f2bf_kernel(const float* __restrict__ s, __bf16* __restrict__ d) {
    int i = blockIdx.x * 256 + threadIdx.x;
    d[i] = (__bf16)s[i];
}

// ---------------- QKV projection GEMM: out[o,ng] = W[o,:] . xnT[ng,:] ----------------
// Software-pipelined: tiles for K-step kb+32 are loaded while step kb's WMMAs run.
// mode 0: Q -> [b][h][n][d] bf16, scaled by 1/sqrt(32); mode 1: K -> [b][h][n][d];
// mode 2: V -> [b][h][d][n]
__global__ void qkv_gemm_kernel(const __bf16* __restrict__ wbf, const __bf16* __restrict__ xnT,
                                const float* __restrict__ bias, __bf16* __restrict__ out, int mode) {
    const int wid  = blockIdx.x * (blockDim.x >> 5) + (threadIdx.x >> 5);
    const int lane = threadIdx.x & 31;
    const int mt = wid >> 7;        // 16 M-tiles of 16 (o)
    const int nq = wid & 127;       // 128 quads of 4 N-tiles
    const int mrow = lane & 15, hi = lane >> 4;
    const int nbase = nq * 64;

    v8f acc0 = {0,0,0,0,0,0,0,0}, acc1 = acc0, acc2 = acc0, acc3 = acc0;
    const __bf16* arow = wbf + (size_t)(mt * 16 + mrow) * C_DIM;
    const __bf16* b0r = xnT + (size_t)(nbase +  0 + mrow) * C_DIM + hi * 16;
    const __bf16* b1r = xnT + (size_t)(nbase + 16 + mrow) * C_DIM + hi * 16;
    const __bf16* b2r = xnT + (size_t)(nbase + 32 + mrow) * C_DIM + hi * 16;
    const __bf16* b3r = xnT + (size_t)(nbase + 48 + mrow) * C_DIM + hi * 16;

    // prologue: load K-step 0
    v16bf Ac = *(const v16bf*)(arow + hi * 16);
    v16bf B0c = *(const v16bf*)(b0r);
    v16bf B1c = *(const v16bf*)(b1r);
    v16bf B2c = *(const v16bf*)(b2r);
    v16bf B3c = *(const v16bf*)(b3r);

    #pragma unroll
    for (int kb = 0; kb < C_DIM; kb += 32) {
        const int kn = (kb + 32) & (C_DIM - 1);   // wraps to 0 on last iter (unused)
        v16bf An = *(const v16bf*)(arow + kn + hi * 16);
        v16bf B0n = *(const v16bf*)(b0r + kn);
        v16bf B1n = *(const v16bf*)(b1r + kn);
        v16bf B2n = *(const v16bf*)(b2r + kn);
        v16bf B3n = *(const v16bf*)(b3r + kn);

        acc0 = wmma_bf16(Ac, B0c, acc0);
        acc1 = wmma_bf16(Ac, B1c, acc1);
        acc2 = wmma_bf16(Ac, B2c, acc2);
        acc3 = wmma_bf16(Ac, B3c, acc3);

        Ac = An; B0c = B0n; B1c = B1n; B2c = B2n; B3c = B3n;
    }

    const float qscale = 0.1767766952966369f;  // 1/sqrt(32)
    v8f accs[4] = {acc0, acc1, acc2, acc3};
    #pragma unroll
    for (int t = 0; t < 4; t++) {
        int ng = nbase + t * 16 + mrow;
        int b = ng >> 12, n = ng & 4095;
        #pragma unroll
        for (int r = 0; r < 8; r++) {
            int o = mt * 16 + hi * 8 + r;    // D layout: row = r + (lane>>4)*8
            float v = accs[t][r] + bias[o];
            int h = o >> 5, d = o & 31;
            if (mode == 2) {
                out[((size_t)((b * HEADS + h) * HD + d)) * NVOX + n] = (__bf16)v;
            } else {
                if (mode == 0) v *= qscale;
                out[((size_t)((b * HEADS + h) * NVOX + n)) * HD + d] = (__bf16)v;
            }
        }
    }
}

// ---------------- Flash attention: one wave per (b,h, 16-row n-tile) ----------------
// K/V tiles for chunk mb+32 are issued before chunk mb's WMMAs, so their latency
// hides behind the softmax VALU work and the LDS P re-stripe.
__global__ void attn_kernel(const __bf16* __restrict__ Qt, const __bf16* __restrict__ Kt,
                            const __bf16* __restrict__ Vm, __bf16* __restrict__ attT) {
    __shared__ __align__(32) __bf16 pt[8][16][32];   // per-wave P re-stripe tile
    const int w    = threadIdx.x >> 5;
    const int wid  = blockIdx.x * 8 + w;
    const int lane = threadIdx.x & 31;
    const int mrow = lane & 15, hi = lane >> 4;
    const int nt = wid & 255;        // 256 row-tiles
    const int bh = wid >> 8;         // b*8+h

    const __bf16* qrow = Qt + ((size_t)bh * NVOX + nt * 16 + mrow) * HD;
    v16bf A = *(const v16bf*)(qrow + hi * 16);
    const __bf16* Kb  = Kt + (size_t)bh * NVOX * HD + (size_t)mrow * HD + hi * 16;
    const __bf16* Kb2 = Kb + (size_t)16 * HD;
    const __bf16* Vb  = Vm + (size_t)bh * HD * NVOX + (size_t)mrow * NVOX + hi * 16;
    const __bf16* Vb2 = Vb + (size_t)16 * NVOX;

    v8f accA = {0,0,0,0,0,0,0,0}, accB = accA;
    const v8f zero = accA;
    float rmax[8], rsum[8];
    #pragma unroll
    for (int r = 0; r < 8; r++) { rmax[r] = -__builtin_inff(); rsum[r] = 0.f; }

    // prologue: chunk 0 tiles
    v16bf KB0c = *(const v16bf*)(Kb);
    v16bf KB1c = *(const v16bf*)(Kb2);
    v16bf VB0c = *(const v16bf*)(Vb);
    v16bf VB1c = *(const v16bf*)(Vb2);

    for (int mb = 0; mb < NVOX; mb += 32) {
        const int mn = (mb + 32) & (NVOX - 1);   // wraps to 0 on last iter (unused)
        v16bf KB0n = *(const v16bf*)(Kb  + (size_t)mn * HD);
        v16bf KB1n = *(const v16bf*)(Kb2 + (size_t)mn * HD);
        v16bf VB0n = *(const v16bf*)(Vb  + mn);
        v16bf VB1n = *(const v16bf*)(Vb2 + mn);

        v8f S0 = wmma_bf16(A, KB0c, zero);
        v8f S1 = wmma_bf16(A, KB1c, zero);

        float cm[8], psum[8];
        #pragma unroll
        for (int r = 0; r < 8; r++) cm[r] = fmaxf(S0[r], S1[r]);
        #pragma unroll
        for (int dd = 1; dd < 16; dd <<= 1) {
            #pragma unroll
            for (int r = 0; r < 8; r++) cm[r] = fmaxf(cm[r], __shfl_xor(cm[r], dd, 32));
        }
        #pragma unroll
        for (int r = 0; r < 8; r++) {
            float nm = fmaxf(rmax[r], cm[r]);
            float corr = __expf(rmax[r] - nm);
            rmax[r] = nm;
            float p0 = __expf(S0[r] - nm), p1 = __expf(S1[r] - nm);
            S0[r] = p0; S1[r] = p1;
            psum[r] = p0 + p1;
            rsum[r] *= corr;
            accA[r] *= corr; accB[r] *= corr;
        }
        #pragma unroll
        for (int dd = 1; dd < 16; dd <<= 1) {
            #pragma unroll
            for (int r = 0; r < 8; r++) psum[r] += __shfl_xor(psum[r], dd, 32);
        }
        #pragma unroll
        for (int r = 0; r < 8; r++) rsum[r] += psum[r];

        // re-stripe P from C/D layout to A layout via LDS (same-wave, in-order DS)
        #pragma unroll
        for (int r = 0; r < 8; r++) {
            pt[w][hi * 8 + r][mrow]      = (__bf16)S0[r];
            pt[w][hi * 8 + r][16 + mrow] = (__bf16)S1[r];
        }
        __asm__ volatile("" ::: "memory");
        v16bf PA = *(const v16bf*)&pt[w][mrow][hi * 16];

        accA = wmma_bf16(PA, VB0c, accA);
        accB = wmma_bf16(PA, VB1c, accB);
        __asm__ volatile("" ::: "memory");

        KB0c = KB0n; KB1c = KB1n; VB0c = VB0n; VB1c = VB1n;
    }

    const int b = bh >> 3, h = bh & 7;
    #pragma unroll
    for (int r = 0; r < 8; r++) {
        float inv = 1.0f / rsum[r];
        int n = nt * 16 + hi * 8 + r;
        size_t rowoff = ((size_t)(b * NVOX + n)) * C_DIM + h * HD;
        attT[rowoff + mrow]      = (__bf16)(accA[r] * inv);
        attT[rowoff + 16 + mrow] = (__bf16)(accB[r] * inv);
    }
}

// ---------------- Output projection + bias + residual (f32 out) ----------------
__global__ void proj_kernel(const __bf16* __restrict__ wob, const __bf16* __restrict__ attT,
                            const float* __restrict__ bo, const float* __restrict__ x,
                            float* __restrict__ out) {
    const int wid  = blockIdx.x * (blockDim.x >> 5) + (threadIdx.x >> 5);
    const int lane = threadIdx.x & 31;
    const int mt = wid >> 7;
    const int nq = wid & 127;
    const int mrow = lane & 15, hi = lane >> 4;
    const int nbase = nq * 64;

    v8f acc0 = {0,0,0,0,0,0,0,0}, acc1 = acc0, acc2 = acc0, acc3 = acc0;
    const __bf16* arow = wob + (size_t)(mt * 16 + mrow) * C_DIM;
    const __bf16* b0r = attT + (size_t)(nbase +  0 + mrow) * C_DIM + hi * 16;
    const __bf16* b1r = attT + (size_t)(nbase + 16 + mrow) * C_DIM + hi * 16;
    const __bf16* b2r = attT + (size_t)(nbase + 32 + mrow) * C_DIM + hi * 16;
    const __bf16* b3r = attT + (size_t)(nbase + 48 + mrow) * C_DIM + hi * 16;

    v16bf Ac = *(const v16bf*)(arow + hi * 16);
    v16bf B0c = *(const v16bf*)(b0r);
    v16bf B1c = *(const v16bf*)(b1r);
    v16bf B2c = *(const v16bf*)(b2r);
    v16bf B3c = *(const v16bf*)(b3r);

    #pragma unroll
    for (int kb = 0; kb < C_DIM; kb += 32) {
        const int kn = (kb + 32) & (C_DIM - 1);
        v16bf An = *(const v16bf*)(arow + kn + hi * 16);
        v16bf B0n = *(const v16bf*)(b0r + kn);
        v16bf B1n = *(const v16bf*)(b1r + kn);
        v16bf B2n = *(const v16bf*)(b2r + kn);
        v16bf B3n = *(const v16bf*)(b3r + kn);

        acc0 = wmma_bf16(Ac, B0c, acc0);
        acc1 = wmma_bf16(Ac, B1c, acc1);
        acc2 = wmma_bf16(Ac, B2c, acc2);
        acc3 = wmma_bf16(Ac, B3c, acc3);

        Ac = An; B0c = B0n; B1c = B1n; B2c = B2n; B3c = B3n;
    }

    v8f accs[4] = {acc0, acc1, acc2, acc3};
    #pragma unroll
    for (int t = 0; t < 4; t++) {
        int ng = nbase + t * 16 + mrow;
        int b = ng >> 12, n = ng & 4095;
        #pragma unroll
        for (int r = 0; r < 8; r++) {
            int o = mt * 16 + hi * 8 + r;
            size_t oi = ((size_t)(b * C_DIM + o)) * NVOX + n;
            out[oi] = x[oi] + accs[t][r] + bo[o];
        }
    }
}

extern "C" void kernel_launch(void* const* d_in, const int* in_sizes, int n_in,
                              void* d_out, int out_size, void* d_ws, size_t ws_size,
                              hipStream_t stream) {
    const float* x    = (const float*)d_in[0];
    const float* gn_w = (const float*)d_in[1];
    const float* gn_b = (const float*)d_in[2];
    const float* wq   = (const float*)d_in[3];
    const float* bq   = (const float*)d_in[4];
    const float* wk   = (const float*)d_in[5];
    const float* bk   = (const float*)d_in[6];
    const float* wv   = (const float*)d_in[7];
    const float* bv   = (const float*)d_in[8];
    const float* wo   = (const float*)d_in[9];
    const float* bo   = (const float*)d_in[10];
    float* out = (float*)d_out;

    char* ws = (char*)d_ws;
    size_t off = 0;
    float* stats = (float*)(ws + off);  off += 256;
    __bf16* xnT  = (__bf16*)(ws + off); off += (size_t)BN * C_DIM * 2;   // 4 MB
    __bf16* wqb  = (__bf16*)(ws + off); off += (size_t)C_DIM * C_DIM * 2;
    __bf16* wkb  = (__bf16*)(ws + off); off += (size_t)C_DIM * C_DIM * 2;
    __bf16* wvb  = (__bf16*)(ws + off); off += (size_t)C_DIM * C_DIM * 2;
    __bf16* wob  = (__bf16*)(ws + off); off += (size_t)C_DIM * C_DIM * 2;
    __bf16* Qt   = (__bf16*)(ws + off); off += (size_t)BN * C_DIM * 2;   // [b][h][n][d]
    __bf16* Kt   = (__bf16*)(ws + off); off += (size_t)BN * C_DIM * 2;   // [b][h][n][d]
    __bf16* Vm   = (__bf16*)(ws + off); off += (size_t)BN * C_DIM * 2;   // [b][h][d][n]
    __bf16* attT = (__bf16*)(ws + off); off += (size_t)BN * C_DIM * 2;   // [b][n][c]

    gn_stats_kernel<<<BATCH * GROUPS, 256, 0, stream>>>(x, stats);
    gn_apply_kernel<<<(BN * C_DIM) / 256, 256, 0, stream>>>(x, stats, gn_w, gn_b, xnT);

    f2bf_kernel<<<(C_DIM * C_DIM) / 256, 256, 0, stream>>>(wq, wqb);
    f2bf_kernel<<<(C_DIM * C_DIM) / 256, 256, 0, stream>>>(wk, wkb);
    f2bf_kernel<<<(C_DIM * C_DIM) / 256, 256, 0, stream>>>(wv, wvb);
    f2bf_kernel<<<(C_DIM * C_DIM) / 256, 256, 0, stream>>>(wo, wob);

    // 2048 waves each: 16 M-tiles x 128 n-quads; 8 waves/block
    qkv_gemm_kernel<<<256, 256, 0, stream>>>(wqb, xnT, bq, Qt, 0);
    qkv_gemm_kernel<<<256, 256, 0, stream>>>(wkb, xnT, bk, Kt, 1);
    qkv_gemm_kernel<<<256, 256, 0, stream>>>(wvb, xnT, bv, Vm, 2);

    // 4096 waves: (b*8+h) x 256 row-tiles; 8 waves/block
    attn_kernel<<<512, 256, 0, stream>>>(Qt, Kt, Vm, attT);

    proj_kernel<<<256, 256, 0, stream>>>(wob, attT, bo, x, out);
}
